// ENAScontroller_21706764714567
// MI455X (gfx1250) — matrix operand order repaired and checked by reading, not compiled, not run on hardware
//
#include <hip/hip_runtime.h>
#include <hip/hip_bf16.h>

typedef __attribute__((ext_vector_type(16))) _Float16 v16h;
typedef __attribute__((ext_vector_type(8)))  float    v8f;

#define DD      256
#define NBR     8
#define NUNITS  12
#define NMAXA   14      // UNITS + 2
#define FTEMP   5.0f
#define FTANHC  1.1f
#define FNEG    (-1.0e9f)
#define NTHR    512     // 16 wave32 waves: ample VGPR budget, no spills

// ---------------------------------------------------------------------------
// Shared-memory state for the whole controller (one workgroup owns everything)
// ---------------------------------------------------------------------------
struct Shm {
  float h[DD], c[DD], x[DD];
  __attribute__((aligned(32))) float g[1024];   // gates; g[0..255] reused as att2h
  __attribute__((aligned(32))) float anchors[NMAXA * DD];
  __attribute__((aligned(32))) float aw1[NMAXA * DD];   // w_attn_1 @ anchor_h
  __attribute__((aligned(32))) float bias[1024];        // b_ih + b_hh
  float wsoft[NBR * DD];
  float vattn[DD];
  float logits[16];
  float bsoft[NBR];
  float lp, ent;
  __attribute__((aligned(32))) _Float16 x16[DD];
  __attribute__((aligned(32))) _Float16 h16[DD];
  int sel;
  unsigned long long rng;
};

__device__ __forceinline__ float sigf(float v) { return 1.0f / (1.0f + expf(-v)); }

// ---------------------------------------------------------------------------
// Pack f32 row-major weights [R x 256] into WMMA A-fragment order (f16).
// Fragment element e of lane L for (tile t, kchunk kc):
//   row = t*16 + (L & 15)
//   k   = kc*32 + ((L >= 16) ? 16 : 0) + e
// a contiguous k order, not the ISA table order; the B vector uses the same one.
// ---------------------------------------------------------------------------
__global__ void enas_pack(const float* __restrict__ src, _Float16* __restrict__ dst,
                          int ntiles) {
  int idx = blockIdx.x * blockDim.x + threadIdx.x;
  int total = ntiles * 8 * 32 * 16;
  if (idx >= total) return;
  int e    = idx & 15;
  int lane = (idx >> 4) & 31;
  int kc   = (idx >> 9) & 7;
  int t    = idx >> 12;
  int row  = t * 16 + (lane & 15);
  int k    = kc * 32 + ((lane >> 4) ? 16 : 0) + e;
  dst[idx] = (_Float16)src[row * DD + k];
}

__device__ __forceinline__ v8f wmma_f16(v16h a, v16h b, v8f c) {
  return __builtin_amdgcn_wmma_f32_16x16x32_f16(false, a, false, b,
                                                (short)0, c, false, false);
}

// ---------------------------------------------------------------------------
// acc += A0[tile] * v0 + A1[tile] * v1   (K = 256 folded as 8 k-chunks,
// interleaved so at most ~4 fragments are live -> no scratch spills).
// B operand: f16 vector replicated across N-columns; lanes 0-15 carry
// K=kc*32+0..15, lanes 16-31 carry K=kc*32+16..31.
// ---------------------------------------------------------------------------
__device__ __forceinline__ v8f mv_dual(const _Float16* __restrict__ pA0,
                                       const _Float16* __restrict__ pA1,
                                       const _Float16* __restrict__ v0,
                                       const _Float16* __restrict__ v1,
                                       int tile, int lane, v8f acc) {
  const size_t fo = ((size_t)tile * 8 * 32 + lane) * 16;   // fragment base
  const _Float16* a0 = pA0 + fo;
  const _Float16* a1 = pA1 + fo;
  const int bo = ((lane >> 4) & 1) * 16;
  const _Float16* b0 = v0 + bo;
  const _Float16* b1 = v1 + bo;
#pragma unroll 2
  for (int kc = 0; kc < 8; ++kc) {
    acc = wmma_f16(*(const v16h*)(a0 + kc * 512), *(const v16h*)(b0 + kc * 32), acc);
    acc = wmma_f16(*(const v16h*)(a1 + kc * 512), *(const v16h*)(b1 + kc * 32), acc);
  }
  return acc;
}

// acc += A[tile] * vec (single matrix variant for the attention matvecs)
__device__ __forceinline__ v8f mv_one(const _Float16* __restrict__ pA,
                                      const _Float16* __restrict__ vec,
                                      int tile, int lane, v8f acc) {
  const _Float16* a = pA + ((size_t)tile * 8 * 32 + lane) * 16;
  const _Float16* b = vec + ((lane >> 4) & 1) * 16;
#pragma unroll 2
  for (int kc = 0; kc < 8; ++kc)
    acc = wmma_f16(*(const v16h*)(a + kc * 512), *(const v16h*)(b + kc * 32), acc);
  return acc;
}

// All 16 columns of the accumulator hold the matvec result; lanes 0 and 16
// write the 16 rows of this tile with one vectorized LDS store each.
__device__ __forceinline__ void mv_store(float* __restrict__ out, int tile, int lane,
                                         v8f acc) {
  if ((lane & 15) == 0) {
    int mb = tile * 16 + ((lane >> 4) ? 8 : 0);
    *(v8f*)(out + mb) = acc;
  }
}

__device__ __forceinline__ void mv_store_bias(float* __restrict__ out,
                                              const float* __restrict__ bias,
                                              int tile, int lane, v8f acc) {
  if ((lane & 15) == 0) {
    int mb = tile * 16 + ((lane >> 4) ? 8 : 0);
    v8f bv = *(const v8f*)(bias + mb);
    *(v8f*)(out + mb) = acc + bv;
  }
}

// ---------------------------------------------------------------------------
// One LSTM cell step: g = W_ih@x + W_hh@h + (b_ih+b_hh); gate math; h,c update.
// 16 waves x 4 tiles = 64 tiles = 1024 gate rows.
// ---------------------------------------------------------------------------
__device__ void lstm_step(Shm* s, const _Float16* pWih, const _Float16* pWhh) {
  const int tid = threadIdx.x, wv = tid >> 5, lane = tid & 31;
  if (tid < DD) { s->x16[tid] = (_Float16)s->x[tid]; s->h16[tid] = (_Float16)s->h[tid]; }
  __syncthreads();
  for (int tt = 0; tt < 4; ++tt) {
    int tile = wv * 4 + tt;
    v8f acc = {0.f, 0.f, 0.f, 0.f, 0.f, 0.f, 0.f, 0.f};
    acc = mv_dual(pWih, pWhh, s->x16, s->h16, tile, lane, acc);
    mv_store_bias(s->g, s->bias, tile, lane, acc);
  }
  __syncthreads();
  if (tid < DD) {
    float gi = s->g[tid], gf = s->g[DD + tid];
    float gc = s->g[2 * DD + tid], go = s->g[3 * DD + tid];
    float cn = sigf(gf) * s->c[tid] + sigf(gi) * tanhf(gc);
    s->c[tid] = cn;
    s->h[tid] = sigf(go) * tanhf(cn);
  }
  __syncthreads();
}

// 256x256 matvec (w_attn_1 or w_attn_2 times h) -> out[0..255]; one tile/wave.
__device__ void mv256(Shm* s, const _Float16* pW, float* out) {
  const int tid = threadIdx.x, wv = tid >> 5, lane = tid & 31;
  if (tid < DD) s->h16[tid] = (_Float16)s->h[tid];
  __syncthreads();
  v8f acc = {0.f, 0.f, 0.f, 0.f, 0.f, 0.f, 0.f, 0.f};
  acc = mv_one(pW, s->h16, wv, lane, acc);
  mv_store(out, wv, lane, acc);
  __syncthreads();
}

// logits[a] = TANHC*tanh( (v_attn . tanh(aw1[a] + att2h)) / TEMP ), masked a<unit
__device__ void attn_logits(Shm* s, int unit) {
  const int tid = threadIdx.x, wv = tid >> 5, lane = tid & 31;
  if (wv < NMAXA) {
    float p = 0.f;
    for (int j = lane; j < DD; j += 32)
      p += s->vattn[j] * tanhf(s->aw1[wv * DD + j] + s->g[j]);
#pragma unroll
    for (int off = 16; off; off >>= 1) p += __shfl_xor(p, off, 32);
    if (lane == 0) {
      float l = FTANHC * tanhf(p / FTEMP);
      s->logits[wv] = (wv < unit) ? l : FNEG;
    }
  }
  __syncthreads();
}

// logits[b] = TANHC*tanh((w_soft[b].h + b_soft[b])/TEMP) (+ b_no_learn if biased)
__device__ void op_logits(Shm* s, bool use_bias) {
  const int tid = threadIdx.x, wv = tid >> 5, lane = tid & 31;
  if (wv < NBR) {
    float p = 0.f;
    for (int j = lane; j < DD; j += 32) p += s->wsoft[wv * DD + j] * s->h[j];
#pragma unroll
    for (int off = 16; off; off >>= 1) p += __shfl_xor(p, off, 32);
    if (lane == 0) {
      float l = FTANHC * tanhf((p + s->bsoft[wv]) / FTEMP);
      if (use_bias) l += (wv < 2 ? 0.25f : -0.25f);
      s->logits[wv] = l;
    }
  }
  __syncthreads();
}

// Categorical sample via inverse CDF + NLL + entropy (thread 0; deterministic RNG).
__device__ void sample(Shm* s, int n) {
  if (threadIdx.x == 0) {
    unsigned long long z = (s->rng += 0x9E3779B97F4A7C15ull);
    z = (z ^ (z >> 30)) * 0xBF58476D1CE4E5B9ull;
    z = (z ^ (z >> 27)) * 0x94D049BB133111EBull;
    z ^= z >> 31;
    float u = (float)(z >> 40) * (1.0f / 16777216.0f);
    float m = FNEG;
    for (int j = 0; j < n; ++j) m = fmaxf(m, s->logits[j]);
    float ps[16], tot = 0.f;
    for (int j = 0; j < n; ++j) { ps[j] = expf(s->logits[j] - m); tot += ps[j]; }
    float target = u * tot, cum = 0.f;
    int idx = n - 1;
    for (int j = 0; j < n; ++j) { cum += ps[j]; if (cum >= target) { idx = j; break; } }
    float logZ = m + logf(tot);
    s->lp += (logZ - s->logits[idx]);           // NLL
    float sl = 0.f;
    for (int j = 0; j < n; ++j) sl += ps[j] * s->logits[j];
    s->ent += (logZ - sl / tot);                // entropy
    s->sel = idx;
  }
  __syncthreads();
}

// ---------------------------------------------------------------------------
// Main persistent kernel: both sampler runs, sequentially, in one workgroup.
// ---------------------------------------------------------------------------
__global__ void __launch_bounds__(NTHR, 1)
enas_main(const float* __restrict__ enc, const float* __restrict__ b_ih,
          const float* __restrict__ b_hh, const float* __restrict__ w_soft,
          const float* __restrict__ b_soft, const float* __restrict__ v_attn,
          const _Float16* __restrict__ pWih, const _Float16* __restrict__ pWhh,
          const _Float16* __restrict__ pW2, const _Float16* __restrict__ pW1,
          float* __restrict__ out) {
  __shared__ Shm s;
  const int tid = threadIdx.x;

  for (int j = tid; j < 1024; j += NTHR) s.bias[j] = b_ih[j] + b_hh[j];
  for (int j = tid; j < NBR * DD; j += NTHR) s.wsoft[j] = w_soft[j];
  if (tid < DD) { s.vattn[tid] = v_attn[tid]; s.h[tid] = 0.f; s.c[tid] = 0.f; }
  if (tid < NBR) s.bsoft[tid] = b_soft[tid];
  if (tid == 0) { s.lp = 0.f; s.ent = 0.f; }
  __syncthreads();

  for (int run = 0; run < 2; ++run) {
    const bool use_bias = (run == 0);
    for (int j = tid; j < NMAXA * DD; j += NTHR) { s.anchors[j] = 0.f; s.aw1[j] = 0.f; }
    if (tid == 0) s.rng = run ? 0xDA3E39CB94B95BDBull : 0x853C49E6748FEA9Bull;
    if (tid < DD) s.x[tid] = enc[tid];                      // encoder[0]
    __syncthreads();

    for (int i = 0; i < 2; ++i) {                           // seed steps
      lstm_step(&s, pWih, pWhh);
      mv256(&s, pW1, &s.aw1[i * DD]);
    }

    for (int unit = 2; unit < 2 + NUNITS; ++unit) {
      const int arcbase = run * 48 + (unit - 2) * 4;
      for (int t = 0; t < 2; ++t) {                         // two input indices
        lstm_step(&s, pWih, pWhh);
        mv256(&s, pW2, s.g);                                // att2h -> g[0..255]
        attn_logits(&s, unit);
        sample(&s, NMAXA);
        if (tid == 0) out[arcbase + t * 2] = (float)s.sel;  // idx slots 0,2
        if (tid < DD) s.x[tid] = s.anchors[s.sel * DD + tid];
        __syncthreads();
      }
      for (int t = 0; t < 2; ++t) {                         // two ops
        lstm_step(&s, pWih, pWhh);
        op_logits(&s, use_bias);
        sample(&s, NBR);
        if (tid == 0) out[arcbase + t * 2 + 1] = (float)s.sel;  // op slots 1,3
        if (tid < DD) s.x[tid] = enc[(s.sel + 1) * DD + tid];
        __syncthreads();
      }
      lstm_step(&s, pWih, pWhh);                            // new anchor
      if (tid < DD) s.anchors[unit * DD + tid] = s.h[tid];
      __syncthreads();
      mv256(&s, pW1, &s.aw1[unit * DD]);
      if (tid < DD) s.x[tid] = enc[tid];
      __syncthreads();
    }
  }
  if (tid == 0) { out[96] = s.lp; out[97] = s.ent; }
}

// ---------------------------------------------------------------------------
extern "C" void kernel_launch(void* const* d_in, const int* in_sizes, int n_in,
                              void* d_out, int out_size, void* d_ws, size_t ws_size,
                              hipStream_t stream) {
  (void)in_sizes; (void)n_in; (void)out_size; (void)ws_size;
  const float* enc    = (const float*)d_in[0];
  const float* W_ih   = (const float*)d_in[1];
  const float* W_hh   = (const float*)d_in[2];
  const float* b_ih   = (const float*)d_in[3];
  const float* b_hh   = (const float*)d_in[4];
  const float* w_soft = (const float*)d_in[5];
  const float* b_soft = (const float*)d_in[6];
  const float* w1     = (const float*)d_in[7];   // w_attn_1
  const float* w2     = (const float*)d_in[8];   // w_attn_2
  const float* v_attn = (const float*)d_in[9];

  const size_t BIG   = (size_t)64 * 8 * 32 * 16;   // 262144 f16 per 1024x256 matrix
  const size_t SMALL = (size_t)16 * 8 * 32 * 16;   //  65536 f16 per  256x256 matrix
  _Float16* pWih = (_Float16*)d_ws;
  _Float16* pWhh = pWih + BIG;
  _Float16* pW2  = pWhh + BIG;
  _Float16* pW1  = pW2 + SMALL;

  enas_pack<<<(int)((BIG + 255) / 256),   256, 0, stream>>>(W_ih, pWih, 64);
  enas_pack<<<(int)((BIG + 255) / 256),   256, 0, stream>>>(W_hh, pWhh, 64);
  enas_pack<<<(int)((SMALL + 255) / 256), 256, 0, stream>>>(w2,   pW2,  16);
  enas_pack<<<(int)((SMALL + 255) / 256), 256, 0, stream>>>(w1,   pW1,  16);

  enas_main<<<1, NTHR, 0, stream>>>(enc, b_ih, b_hh, w_soft, b_soft, v_attn,
                                    pWih, pWhh, pW2, pW1, (float*)d_out);
}
